// FusedRobertaLayerIntraInter_80908593922051
// MI455X (gfx1250) — matrix operand order repaired and checked by reading, not compiled, not run on hardware
//
#include <hip/hip_runtime.h>

// ---------------------------------------------------------------------------
// Problem constants (B,S,D,H,HD,F,R from the reference)
// ---------------------------------------------------------------------------
constexpr int CB  = 16;          // batch
constexpr int CS  = 512;         // seq len
constexpr int CD  = 1024;        // hidden
constexpr int CH  = 16;          // heads
constexpr int CHD = 64;          // head dim
constexpr int CF  = 4096;        // ffn dim
constexpr int CR  = 16;          // lora rank
constexpr int NTOK = CB * CS;    // 8192 tokens

typedef __attribute__((ext_vector_type(16))) __bf16 v16bf;
typedef __attribute__((ext_vector_type(8)))  __bf16 v8bf;
typedef __attribute__((ext_vector_type(8)))  float  v8f;
typedef __attribute__((ext_vector_type(4)))  unsigned int u32x4;
typedef __attribute__((ext_vector_type(8)))  int i32x8;
typedef __attribute__((ext_vector_type(4)))  int i32x4;

union Frag16 { v16bf v; v8bf h8[2]; };

// ---------------------------------------------------------------------------
// Tensor Data Mover availability (builtin exists on both toolchains, but with
// different arity: 5 args on ROCm7.2/clang-22, 6 args on clang-23+).
// ---------------------------------------------------------------------------
#if defined(__has_builtin)
#  if __has_builtin(__builtin_amdgcn_tensor_load_to_lds)
#    define HAVE_TDM 1
#  endif
#endif
#ifndef HAVE_TDM
#  define HAVE_TDM 0
#endif

#if HAVE_TDM
// 2D TDM load: tile (tile_d1 rows x tile_d0 elems) of a (tensor_d1 x tensor_d0)
// bf16 tensor with row stride `stride0` (elements) -> LDS at lds_addr.
// D# packing per CDNA5 ISA.
__device__ __forceinline__ void tdm_load_2d(unsigned lds_addr, const void* gaddr,
                                            unsigned tensor_d0, unsigned tensor_d1,
                                            unsigned tile_d0, unsigned tile_d1,
                                            unsigned stride0) {
    unsigned long long ga = (unsigned long long)(uintptr_t)gaddr;
    u32x4 g0;
    g0[0] = 1u;                                   // count=1 (valid), user mode
    g0[1] = lds_addr;                             // lds_addr [63:32]
    g0[2] = (unsigned)ga;                         // global_addr[31:0]
    g0[3] = (unsigned)((ga >> 32) & 0x01FFFFFFu)  // global_addr[56:32]
          | (2u << 30);                           // type=2 ("image")
    i32x8 g1;
    g1[0] = (int)(1u << 16);                      // wg_mask=0, data_size=1 (2B)
    g1[1] = (int)((tensor_d0 & 0xFFFFu) << 16);   // tensor_dim0[15:0] @ 63:48
    g1[2] = (int)(((tensor_d0 >> 16) & 0xFFFFu)   // tensor_dim0[31:16]
          | ((tensor_d1 & 0xFFFFu) << 16));       // tensor_dim1[15:0]
    g1[3] = (int)(((tensor_d1 >> 16) & 0xFFFFu)   // tensor_dim1[31:16]
          | (tile_d0 << 16));                     // tile_dim0
    g1[4] = (int)(tile_d1 & 0xFFFFu);             // tile_dim1 (tile_dim2=0)
    g1[5] = (int)stride0;                         // tensor_dim0_stride[31:0]
    g1[6] = 0;                                    // stride0[47:32]=0, stride1=0
    g1[7] = 0;
    i32x4 gz = {0, 0, 0, 0};
#if __clang_major__ >= 23
    i32x8 gz8 = {0, 0, 0, 0, 0, 0, 0, 0};
    __builtin_amdgcn_tensor_load_to_lds(g0, g1, gz, gz, gz8, 0);
#else
    __builtin_amdgcn_tensor_load_to_lds(g0, g1, gz, gz, 0);
#endif
}
#endif

// ---------------------------------------------------------------------------
// fp32 -> bf16 conversion (activations)
// ---------------------------------------------------------------------------
__global__ void cvt_f32_bf16_kernel(const float* __restrict__ src,
                                    __bf16* __restrict__ dst, size_t n) {
    size_t i = blockIdx.x * (size_t)blockDim.x + threadIdx.x;
    size_t stride = gridDim.x * (size_t)blockDim.x;
    for (; i < n; i += stride) dst[i] = (__bf16)src[i];
}

// ---------------------------------------------------------------------------
// fp32 [K x N] -> bf16 transposed [N x K] (weights).  32x32 tiles via LDS.
// ---------------------------------------------------------------------------
__global__ void cvt_transpose_kernel(const float* __restrict__ src,
                                     __bf16* __restrict__ dst, int K, int N) {
    __shared__ float tile[32][33];
    int n0 = blockIdx.x * 32, k0 = blockIdx.y * 32;
    int tx = threadIdx.x & 31, ty = threadIdx.x >> 5;   // 32 x 8
    for (int i = ty; i < 32; i += 8)
        tile[i][tx] = src[(size_t)(k0 + i) * N + n0 + tx];
    __syncthreads();
    for (int i = ty; i < 32; i += 8)
        dst[(size_t)(n0 + i) * K + k0 + tx] = (__bf16)tile[tx][i];
}

// ---------------------------------------------------------------------------
// LoRA first stage: t[M x 16] = act(bf16, M x K) @ a(f32, K x 16)
// ---------------------------------------------------------------------------
__global__ void lora_t_kernel(const __bf16* __restrict__ act,
                              const float* __restrict__ a,
                              float* __restrict__ t, int Kdim) {
    int m = blockIdx.x * 16 + threadIdx.y;
    int r = threadIdx.x;                       // 0..15
    const __bf16* ap = act + (size_t)m * Kdim;
    float s = 0.f;
    for (int k = 0; k < Kdim; k += 8) {
        v8bf av = *(const v8bf*)(ap + k);
#pragma unroll
        for (int i = 0; i < 8; ++i)
            s += (float)av[i] * a[(size_t)(k + i) * CR + r];
    }
    t[(size_t)m * CR + r] = s;
}

// ---------------------------------------------------------------------------
// Tiled WMMA bf16 GEMM:  C[M x N] = A(bf16, MxK) @ BwT(bf16, NxK transposed)
// BM=128, BN=64, BK=32.  256 threads = 8 waves, wave grid 4x2, 32x32/wave.
// With TDM: both tiles are plain row-major 2D tiles fetched by the Tensor
// Data Mover into double-buffered LDS; pipeline uses s_wait_tensorcnt.
// Fragment layouts per ISA: A 16x32 bf16 (two contiguous 16B chunks/lane),
// B column (lane&15) contiguous because weights are pre-transposed.
// ---------------------------------------------------------------------------
template<bool LORA, bool BIAS, bool RESID, bool GELU, bool OUTF, bool OUTB>
__global__ void gemm_bf16_kernel(const __bf16* __restrict__ A,
                                 const __bf16* __restrict__ BwT,  // N x K
                                 const float*  __restrict__ bias,
                                 const float*  __restrict__ lt,   // M x 16
                                 const float*  __restrict__ lb,   // 16 x N
                                 const float*  __restrict__ resid,
                                 float*        __restrict__ Cf,
                                 __bf16*       __restrict__ Cb,
                                 int M, int N, int K) {
    constexpr int BM = 128, BN = 64, BK = 32;
#if HAVE_TDM
    constexpr int NBUF = 2;
#else
    constexpr int NBUF = 1;
#endif
    __shared__ __bf16 sA[NBUF * BM * BK];   // row-major, stride BK
    __shared__ __bf16 sB[NBUF * BN * BK];   // row-major [n][k], stride BK

    const int tid  = threadIdx.x;
    const int lane = tid & 31;
    const int w    = tid >> 5;
    const int wm   = (w >> 1) * 32;
    const int wn   = (w & 1) * 32;
    const int bm   = blockIdx.y * BM;
    const int bn   = blockIdx.x * BN;
    const int half = lane >> 4;
    const int l16  = lane & 15;
    const int steps = K / BK;

    v8f acc[2][2] = {};

#if HAVE_TDM
    if (tid < 32) {   // wave 0 drives the TDM (EXEC is ignored by tensor ops)
        tdm_load_2d((unsigned)(uintptr_t)&sA[0], A + (size_t)bm * K,
                    (unsigned)K, (unsigned)M, BK, BM, (unsigned)K);
        tdm_load_2d((unsigned)(uintptr_t)&sB[0], BwT + (size_t)bn * K,
                    (unsigned)K, (unsigned)N, BK, BN, (unsigned)K);
    }
#endif

    for (int s = 0; s < steps; ++s) {
        const int bsel = s & (NBUF - 1);
        const __bf16* cA = &sA[bsel * BM * BK];
        const __bf16* cB = &sB[bsel * BN * BK];

#if HAVE_TDM
        if (tid < 32) {
            if (s + 1 < steps) {   // prefetch next tile pair into other buffer
                const int nb = (s + 1) & 1;
                tdm_load_2d((unsigned)(uintptr_t)&sA[nb * BM * BK],
                            A + (size_t)bm * K + (s + 1) * BK,
                            (unsigned)K, (unsigned)M, BK, BM, (unsigned)K);
                tdm_load_2d((unsigned)(uintptr_t)&sB[nb * BN * BK],
                            BwT + (size_t)bn * K + (s + 1) * BK,
                            (unsigned)K, (unsigned)N, BK, BN, (unsigned)K);
                __builtin_amdgcn_s_wait_tensorcnt(2);  // current pair landed
            } else {
                __builtin_amdgcn_s_wait_tensorcnt(0);
            }
        }
        __syncthreads();
#else
        {   // manual staging fallback
            const int k0 = s * BK;
            int r = tid >> 1;
            int c = (tid & 1) * 16;
            const __bf16* ga = A + (size_t)(bm + r) * K + k0 + c;
            *(v8bf*)&sA[r * BK + c]     = *(const v8bf*)(ga);
            *(v8bf*)&sA[r * BK + c + 8] = *(const v8bf*)(ga + 8);
            int rn = tid >> 2;            // 0..63
            int cn = (tid & 3) * 8;       // 0..24
            *(v8bf*)&sB[rn * BK + cn] =
                *(const v8bf*)(BwT + (size_t)(bn + rn) * K + k0 + cn);
            __syncthreads();
            if (k0 + BK < K) {
                __builtin_prefetch(A + (size_t)(bm + r) * K + k0 + BK, 0, 1);
                __builtin_prefetch(BwT + (size_t)(bn + rn) * K + k0 + BK, 0, 1);
            }
        }
#endif

#pragma unroll
        for (int mi = 0; mi < 2; ++mi) {
            Frag16 af;
            const __bf16* pa = &cA[(wm + mi * 16 + l16) * BK];
            af.h8[0] = *(const v8bf*)(pa + half * 8);
            af.h8[1] = *(const v8bf*)(pa + 16 + half * 8);
#pragma unroll
            for (int ni = 0; ni < 2; ++ni) {
                Frag16 bf;
                const __bf16* pb = &cB[(wn + ni * 16 + l16) * BK + half * 8];
                bf.h8[0] = *(const v8bf*)(pb);
                bf.h8[1] = *(const v8bf*)(pb + 16);
                acc[mi][ni] = __builtin_amdgcn_wmma_f32_16x16x32_bf16(
                    false, af.v, false, bf.v, (short)0, acc[mi][ni], false, false);
            }
        }
        __syncthreads();
    }

    // ---- epilogue ----
#pragma unroll
    for (int mi = 0; mi < 2; ++mi) {
#pragma unroll
        for (int ni = 0; ni < 2; ++ni) {
#pragma unroll
            for (int r = 0; r < 8; ++r) {
                int row = bm + wm + mi * 16 + half * 8 + r;
                int col = bn + wn + ni * 16 + l16;
                float v = acc[mi][ni][r];
                if (BIAS) v += bias[col];
                if (LORA) {
                    const float* tp = lt + (size_t)row * CR;
                    const float* bp = lb + col;
#pragma unroll
                    for (int j = 0; j < CR; ++j) v += tp[j] * bp[(size_t)j * N];
                }
                if (RESID) v += resid[(size_t)row * N + col];
                if (GELU)  v = 0.5f * v * (1.f + erff(v * 0.70710678118f));
                if (OUTF) Cf[(size_t)row * N + col] = v;
                if (OUTB) Cb[(size_t)row * N + col] = (__bf16)v;
            }
        }
    }
}

// ---------------------------------------------------------------------------
// Flash attention (online softmax) with WMMA for QK^T and P@V.
// grid = (S/128, H, B); block = 256 = 8 waves; each wave owns 16 query rows.
// ---------------------------------------------------------------------------
__global__ void flash_attn_kernel(const __bf16* __restrict__ Q,
                                  const __bf16* __restrict__ Kb,
                                  const __bf16* __restrict__ Vb,
                                  const float*  __restrict__ mask,
                                  __bf16*       __restrict__ Ob) {
    const int h = blockIdx.y, b = blockIdx.z;
    const int qbase = b * CS + blockIdx.x * 128;
    const int tid  = threadIdx.x;
    const int lane = tid & 31;
    const int w    = tid >> 5;
    const int half = lane >> 4;
    const int l16  = lane & 15;

    __shared__ __bf16 sK [32 * 64];        // [kv][hd]
    __shared__ __bf16 sVT[64 * 32];        // [hd][kv]  (transposed V)
    __shared__ __bf16 sP [8][16 * 32];     // per-wave P scratch

    const int qrow = qbase + w * 16 + l16;            // A row = l16
    const __bf16* qp = Q + (size_t)qrow * CD + h * CHD;
    Frag16 qa[2];
#pragma unroll
    for (int s = 0; s < 2; ++s) {
        qa[s].h8[0] = *(const v8bf*)(qp + s * 32 + half * 8);
        qa[s].h8[1] = *(const v8bf*)(qp + s * 32 + 16 + half * 8);
    }

    v8f o[4] = {};
    float m_[8], l_[8];
#pragma unroll
    for (int r = 0; r < 8; ++r) { m_[r] = -1e30f; l_[r] = 0.f; }

    const float scale = 0.125f;    // 1/sqrt(64)

    for (int j0 = 0; j0 < CS; j0 += 32) {
        __syncthreads();
        {
            int kv = tid >> 3;
            int c  = (tid & 7) * 8;
            const __bf16* gk = Kb + (size_t)(b * CS + j0 + kv) * CD + h * CHD + c;
            *(v8bf*)&sK[kv * 64 + c] = *(const v8bf*)gk;
            const __bf16* gv = Vb + (size_t)(b * CS + j0 + kv) * CD + h * CHD + c;
            v8bf vv = *(const v8bf*)gv;
#pragma unroll
            for (int i = 0; i < 8; ++i) sVT[(c + i) * 32 + kv] = vv[i];
        }
        __syncthreads();

        v8f sc[2] = {};
#pragma unroll
        for (int ni = 0; ni < 2; ++ni) {
#pragma unroll
            for (int s = 0; s < 2; ++s) {
                Frag16 kb;
                const __bf16* pk = &sK[(ni * 16 + l16) * 64 + s * 32 + half * 8];
                kb.h8[0] = *(const v8bf*)(pk);
                kb.h8[1] = *(const v8bf*)(pk + 16);
                sc[ni] = __builtin_amdgcn_wmma_f32_16x16x32_bf16(
                    false, qa[s].v, false, kb.v, (short)0, sc[ni], false, false);
            }
        }

        float mk0 = mask[(size_t)b * CS + j0 + l16];
        float mk1 = mask[(size_t)b * CS + j0 + 16 + l16];
#pragma unroll
        for (int r = 0; r < 8; ++r) {
            float t0 = sc[0][r] * scale + mk0;
            float t1 = sc[1][r] * scale + mk1;
            float mx = fmaxf(t0, t1);
#pragma unroll
            for (int d = 1; d < 16; d <<= 1) mx = fmaxf(mx, __shfl_xor(mx, d, 32));
            float mnew = fmaxf(m_[r], mx);
            float corr = __expf(m_[r] - mnew);
            float p0 = __expf(t0 - mnew);
            float p1 = __expf(t1 - mnew);
            float rs = p0 + p1;
#pragma unroll
            for (int d = 1; d < 16; d <<= 1) rs += __shfl_xor(rs, d, 32);
            l_[r] = l_[r] * corr + rs;
            m_[r] = mnew;
            sc[0][r] = p0;
            sc[1][r] = p1;
#pragma unroll
            for (int t = 0; t < 4; ++t) o[t][r] *= corr;
        }

#pragma unroll
        for (int ni = 0; ni < 2; ++ni)
#pragma unroll
            for (int r = 0; r < 8; ++r)
                sP[w][(half * 8 + r) * 32 + ni * 16 + l16] = (__bf16)sc[ni][r];
        __builtin_amdgcn_wave_barrier();   // LDS ops are wave-ordered

        Frag16 pa;
        const __bf16* pp = &sP[w][l16 * 32];
        pa.h8[0] = *(const v8bf*)(pp + half * 8);
        pa.h8[1] = *(const v8bf*)(pp + 16 + half * 8);

#pragma unroll
        for (int t = 0; t < 4; ++t) {
            Frag16 vb;
            const __bf16* pv = &sVT[(t * 16 + l16) * 32 + half * 8];
            vb.h8[0] = *(const v8bf*)(pv);
            vb.h8[1] = *(const v8bf*)(pv + 16);
            o[t] = __builtin_amdgcn_wmma_f32_16x16x32_bf16(
                false, pa.v, false, vb.v, (short)0, o[t], false, false);
        }
    }

#pragma unroll
    for (int t = 0; t < 4; ++t)
#pragma unroll
        for (int r = 0; r < 8; ++r) {
            int token = qbase + w * 16 + half * 8 + r;
            int col   = h * CHD + t * 16 + l16;
            Ob[(size_t)token * CD + col] = (__bf16)(o[t][r] / l_[r]);
        }
}

// ---------------------------------------------------------------------------
// LayerNorm over D=1024; one block (256 threads) per token.
// ---------------------------------------------------------------------------
__global__ void layernorm_kernel(const float* __restrict__ in,
                                 const float* __restrict__ g,
                                 const float* __restrict__ be,
                                 float*  __restrict__ outF,
                                 __bf16* __restrict__ outB) {
    const int row = blockIdx.x;
    const float* p = in + (size_t)row * CD;
    float s = 0.f, s2 = 0.f;
    for (int i = threadIdx.x; i < CD; i += 256) {
        float v = p[i]; s += v; s2 += v * v;
    }
#pragma unroll
    for (int d = 1; d < 32; d <<= 1) {
        s  += __shfl_xor(s,  d, 32);
        s2 += __shfl_xor(s2, d, 32);
    }
    __shared__ float sw[8], sw2[8];
    int w = threadIdx.x >> 5, lane = threadIdx.x & 31;
    if (lane == 0) { sw[w] = s; sw2[w] = s2; }
    __syncthreads();
    s = 0.f; s2 = 0.f;
#pragma unroll
    for (int i = 0; i < 8; ++i) { s += sw[i]; s2 += sw2[i]; }
    float mu  = s * (1.f / CD);
    float var = s2 * (1.f / CD) - mu * mu;
    float inv = rsqrtf(var + 1e-5f);
    for (int i = threadIdx.x; i < CD; i += 256) {
        float v = (p[i] - mu) * inv * g[i] + be[i];
        if (outF) outF[(size_t)row * CD + i] = v;
        if (outB) outB[(size_t)row * CD + i] = (__bf16)v;
    }
}

// ---------------------------------------------------------------------------
// Host-side orchestration
// ---------------------------------------------------------------------------
extern "C" void kernel_launch(void* const* d_in, const int* in_sizes, int n_in,
                              void* d_out, int out_size, void* d_ws, size_t ws_size,
                              hipStream_t stream) {
    const float* x    = (const float*)d_in[0];
    const float* w_q  = (const float*)d_in[1];
    const float* b_q  = (const float*)d_in[2];
    const float* aq   = (const float*)d_in[3];
    const float* bq   = (const float*)d_in[4];
    const float* w_k  = (const float*)d_in[5];
    const float* b_k  = (const float*)d_in[6];
    const float* ak   = (const float*)d_in[7];
    const float* bk   = (const float*)d_in[8];
    const float* w_v  = (const float*)d_in[9];
    const float* b_v  = (const float*)d_in[10];
    const float* av   = (const float*)d_in[11];
    const float* bv   = (const float*)d_in[12];
    const float* w_o  = (const float*)d_in[13];
    const float* b_o  = (const float*)d_in[14];
    const float* ao   = (const float*)d_in[15];
    const float* bo   = (const float*)d_in[16];
    const float* nw1  = (const float*)d_in[17];
    const float* nb1  = (const float*)d_in[18];
    const float* w_up = (const float*)d_in[19];
    const float* b_up = (const float*)d_in[20];
    const float* aup  = (const float*)d_in[21];
    const float* bup  = (const float*)d_in[22];
    const float* w_dn = (const float*)d_in[23];
    const float* b_dn = (const float*)d_in[24];
    const float* adn  = (const float*)d_in[25];
    const float* bdn  = (const float*)d_in[26];
    const float* nw2  = (const float*)d_in[27];
    const float* nb2  = (const float*)d_in[28];
    const float* mask = (const float*)d_in[29];
    float* out        = (float*)d_out;

    char* p = (char*)d_ws;
    size_t used = 0;
    auto alloc = [&](size_t bytes) {
        char* r = p + used;
        used += (bytes + 255) & ~(size_t)255;
        return r;
    };
    // weights stored TRANSPOSED (N x K) in bf16
    __bf16* wqT  = (__bf16*)alloc((size_t)CD * CD * 2);
    __bf16* wkT  = (__bf16*)alloc((size_t)CD * CD * 2);
    __bf16* wvT  = (__bf16*)alloc((size_t)CD * CD * 2);
    __bf16* woT  = (__bf16*)alloc((size_t)CD * CD * 2);
    __bf16* wupT = (__bf16*)alloc((size_t)CD * CF * 2);
    __bf16* wdnT = (__bf16*)alloc((size_t)CF * CD * 2);
    __bf16* x_bf = (__bf16*)alloc((size_t)NTOK * CD * 2);
    __bf16* q_bf = (__bf16*)alloc((size_t)NTOK * CD * 2);
    __bf16* k_bf = (__bf16*)alloc((size_t)NTOK * CD * 2);
    __bf16* v_bf = (__bf16*)alloc((size_t)NTOK * CD * 2);
    __bf16* o_bf = (__bf16*)alloc((size_t)NTOK * CD * 2);
    __bf16* xm_bf= (__bf16*)alloc((size_t)NTOK * CD * 2);
    __bf16* fn_bf= (__bf16*)alloc((size_t)NTOK * CF * 2);
    float*  t_ws = (float*) alloc((size_t)NTOK * CR * 4);
    float*  ofin = (float*) alloc((size_t)NTOK * CD * 4);
    float*  xm_f = (float*) alloc((size_t)NTOK * CD * 4);
    float*  dn_f = (float*) alloc((size_t)NTOK * CD * 4);
    if (used > ws_size) return;   // insufficient scratch: do nothing (safe)

    const dim3 blk256(256);
    // 1. weight transpose+convert (K x N f32 -> N x K bf16), x convert
    cvt_transpose_kernel<<<dim3(CD / 32, CD / 32), blk256, 0, stream>>>(w_q,  wqT,  CD, CD);
    cvt_transpose_kernel<<<dim3(CD / 32, CD / 32), blk256, 0, stream>>>(w_k,  wkT,  CD, CD);
    cvt_transpose_kernel<<<dim3(CD / 32, CD / 32), blk256, 0, stream>>>(w_v,  wvT,  CD, CD);
    cvt_transpose_kernel<<<dim3(CD / 32, CD / 32), blk256, 0, stream>>>(w_o,  woT,  CD, CD);
    cvt_transpose_kernel<<<dim3(CF / 32, CD / 32), blk256, 0, stream>>>(w_up, wupT, CD, CF);
    cvt_transpose_kernel<<<dim3(CD / 32, CF / 32), blk256, 0, stream>>>(w_dn, wdnT, CF, CD);
    cvt_f32_bf16_kernel<<<2048, blk256, 0, stream>>>(x, x_bf, (size_t)NTOK * CD);

    const dim3 loraBlk(16, 16);
    const dim3 loraGrd(NTOK / 16);
    const dim3 gemmGrdD(CD / 64, NTOK / 128);   // N=1024
    const dim3 gemmGrdF(CF / 64, NTOK / 128);   // N=4096

    // 2-4. Q, K, V projections (+bias +lora) -> bf16
    lora_t_kernel<<<loraGrd, loraBlk, 0, stream>>>(x_bf, aq, t_ws, CD);
    gemm_bf16_kernel<true, true, false, false, false, true>
        <<<gemmGrdD, blk256, 0, stream>>>(x_bf, wqT, b_q, t_ws, bq,
                                          nullptr, nullptr, q_bf, NTOK, CD, CD);
    lora_t_kernel<<<loraGrd, loraBlk, 0, stream>>>(x_bf, ak, t_ws, CD);
    gemm_bf16_kernel<true, true, false, false, false, true>
        <<<gemmGrdD, blk256, 0, stream>>>(x_bf, wkT, b_k, t_ws, bk,
                                          nullptr, nullptr, k_bf, NTOK, CD, CD);
    lora_t_kernel<<<loraGrd, loraBlk, 0, stream>>>(x_bf, av, t_ws, CD);
    gemm_bf16_kernel<true, true, false, false, false, true>
        <<<gemmGrdD, blk256, 0, stream>>>(x_bf, wvT, b_v, t_ws, bv,
                                          nullptr, nullptr, v_bf, NTOK, CD, CD);

    // 5. flash attention
    flash_attn_kernel<<<dim3(CS / 128, CH, CB), blk256, 0, stream>>>(
        q_bf, k_bf, v_bf, mask, o_bf);

    // 6. output projection + residual(x) -> ofin (f32)
    lora_t_kernel<<<loraGrd, loraBlk, 0, stream>>>(o_bf, ao, t_ws, CD);
    gemm_bf16_kernel<true, true, true, false, true, false>
        <<<gemmGrdD, blk256, 0, stream>>>(o_bf, woT, b_o, t_ws, bo,
                                          x, ofin, nullptr, NTOK, CD, CD);

    // 7. layernorm 1 -> xm_f (f32) + xm_bf (bf16)
    layernorm_kernel<<<NTOK, blk256, 0, stream>>>(ofin, nw1, nb1, xm_f, xm_bf);

    // 8. up projection + GELU fused -> fn_bf
    lora_t_kernel<<<loraGrd, loraBlk, 0, stream>>>(xm_bf, aup, t_ws, CD);
    gemm_bf16_kernel<true, true, false, true, false, true>
        <<<gemmGrdF, blk256, 0, stream>>>(xm_bf, wupT, b_up, t_ws, bup,
                                          nullptr, nullptr, fn_bf, NTOK, CF, CD);

    // 9. down projection + residual(xm_f) -> dn_f
    lora_t_kernel<<<loraGrd, loraBlk, 0, stream>>>(fn_bf, adn, t_ws, CF);
    gemm_bf16_kernel<true, true, true, false, true, false>
        <<<gemmGrdD, blk256, 0, stream>>>(fn_bf, wdnT, b_dn, t_ws, bdn,
                                          xm_f, dn_f, nullptr, NTOK, CD, CF);

    // 10. layernorm 2 -> final output (f32)
    layernorm_kernel<<<NTOK, blk256, 0, stream>>>(dn_f, nw2, nb2, out, nullptr);
}
